// VDGNGATStep_46686294507706
// MI455X (gfx1250) — matrix operand order repaired and checked by reading, not compiled, not run on hardware
//
#include <hip/hip_runtime.h>
#include <hip/hip_bf16.h>

typedef __attribute__((ext_vector_type(16))) _Float16 v16h;
typedef __attribute__((ext_vector_type(8)))  float    v8f;

#define WMMA_F16(a, b, c) \
  __builtin_amdgcn_wmma_f32_16x16x32_f16(false, (a), false, (b), (short)0, (c), false, false)

// ---------------- helpers ----------------

__device__ __forceinline__ unsigned encf(float f) {
  int i = __float_as_int(f);
  return (i >= 0) ? ((unsigned)i | 0x80000000u) : ~(unsigned)i;
}
__device__ __forceinline__ float decf(unsigned u) {
  int i = (u & 0x80000000u) ? (int)(u & 0x7fffffffu) : (int)~u;
  return __int_as_float(i);
}

// Pre-swizzle a 128x128 f32 weight matrix (row-major, K x N) into LDS as f16
// WMMA B-fragments. Fragment f = ks*8 + nt (ks = K/32 step, nt = N/16 tile).
// Half i of lane l in fragment f holds B[ks*32 + 8*(l>=16) + (i&7) + 16*(i>=8)][nt*16 + (l&15)].
__device__ __forceinline__ void fill_wfrag(const float* __restrict__ W, _Float16* lds) {
  for (int t = threadIdx.x; t < 128 * 128; t += blockDim.x) {
    int k = t >> 7, n = t & 127;
    int ks = k >> 5, kr = k & 31;
    int hi = (kr >> 3) & 1, i = (kr & 7) | ((kr >> 4) << 3);
    int lane = (n & 15) + (hi << 4);
    int f = ks * 8 + (n >> 4);
    lds[(f * 32 + lane) * 16 + i] = (_Float16)W[t];
  }
}

__device__ __forceinline__ v16h load_bfrag(const _Float16* lds, int f, int lane) {
  union { v16h v; uint4 u[2]; } r;
  const uint4* p = (const uint4*)(lds + ((f * 32 + lane) << 4));
  r.u[0] = p[0];
  r.u[1] = p[1];
  return r.v;
}

// f16 A-fragment (16 rows x 32 K) from a row-major f32 matrix with ld=128.
// Per ISA: lanes 0-15 row m=lane hold K = kb+0..7 (halfs 0-7) and kb+16..23
// (halfs 8-15); lanes 16-31 hold kb+8..15 and kb+24..31. Raw f32 values are
// also returned for the attention-fold accumulation.
__device__ __forceinline__ v16h load_afrag(const float* __restrict__ A, long long row0,
                                           int kbase, int lane, float av[16]) {
  long long r = row0 + (lane & 15);
  int kh = kbase + ((lane >> 4) << 3);
  const float* base = A + r * 128 + kh;
  float4 q0 = *(const float4*)(base + 0);
  float4 q1 = *(const float4*)(base + 4);
  float4 q2 = *(const float4*)(base + 16);
  float4 q3 = *(const float4*)(base + 20);
  av[0] = q0.x;  av[1] = q0.y;  av[2] = q0.z;  av[3] = q0.w;
  av[4] = q1.x;  av[5] = q1.y;  av[6] = q1.z;  av[7] = q1.w;
  av[8] = q2.x;  av[9] = q2.y;  av[10] = q2.z; av[11] = q2.w;
  av[12] = q3.x; av[13] = q3.y; av[14] = q3.z; av[15] = q3.w;
  union { v16h v; _Float16 h[16]; } c;
#pragma unroll
  for (int i = 0; i < 16; i++) c.h[i] = (_Float16)av[i];
  return c.v;
}

__device__ __forceinline__ v16h load_afrag_lds(const float* tb, int kbase, int lane) {
  int r = lane & 15;
  int kh = kbase + ((lane >> 4) << 3);
  const float* base = tb + r * 128 + kh;
  float4 q0 = *(const float4*)(base + 0);
  float4 q1 = *(const float4*)(base + 4);
  float4 q2 = *(const float4*)(base + 16);
  float4 q3 = *(const float4*)(base + 20);
  float av[16] = {q0.x, q0.y, q0.z, q0.w, q1.x, q1.y, q1.z, q1.w,
                  q2.x, q2.y, q2.z, q2.w, q3.x, q3.y, q3.z, q3.w};
  union { v16h v; _Float16 h[16]; } c;
#pragma unroll
  for (int i = 0; i < 16; i++) c.h[i] = (_Float16)av[i];
  return c.v;
}

// ---------------- K0: fold attention vectors through the weights ----------------
// v_src[d][h] = sum_c W_node[d][h*64+c]*att_src[h][c]   (same for dst / edge)
__global__ __launch_bounds__(128) void k_fold(
    const float* __restrict__ Wn, const float* __restrict__ as_,
    const float* __restrict__ ad_, const float* __restrict__ We,
    const float* __restrict__ ae_, float* __restrict__ vsrc,
    float* __restrict__ vdst, float* __restrict__ vedge) {
  int d = threadIdx.x;
#pragma unroll
  for (int h = 0; h < 2; h++) {
    float s = 0.f, t = 0.f, u = 0.f;
    for (int c = 0; c < 64; c++) {
      float wn = Wn[d * 128 + h * 64 + c];
      float we = We[d * 128 + h * 64 + c];
      s += wn * as_[h * 64 + c];
      t += wn * ad_[h * 64 + c];
      u += we * ae_[h * 64 + c];
    }
    vsrc[d * 2 + h] = s;
    vdst[d * 2 + h] = t;
    vedge[d * 2 + h] = u;
  }
}

// ---------------- K_init: zero the accumulators (every launch) ----------------
__global__ __launch_bounds__(256) void k_init(float* __restrict__ outacc,
                                              unsigned* __restrict__ amaxK,
                                              float* __restrict__ denom,
                                              long long n128, long long n2) {
  long long i = (long long)blockIdx.x * blockDim.x + threadIdx.x;
  if (i < n128) outacc[i] = 0.f;
  if (i < n2) { amaxK[i] = 0u; denom[i] = 0.f; }
}

// ---------------- K1: xh = x @ W_node (WMMA f16) + folded a_src/a_dst ----------------
// Grid-stride over 64-row tiles so the LDS weight fill amortizes.
__global__ __launch_bounds__(128) void k_node_gemm(
    const float* __restrict__ x, const float* __restrict__ Wn,
    const float* __restrict__ vsrc, const float* __restrict__ vdst,
    float* __restrict__ xh, float* __restrict__ asrc, float* __restrict__ adst,
    long long Nn) {
  __shared__ _Float16 wfrag[16384];
  __shared__ float vs[256], vd[256];
  fill_wfrag(Wn, wfrag);
  for (int t = threadIdx.x; t < 256; t += blockDim.x) { vs[t] = vsrc[t]; vd[t] = vdst[t]; }
  __syncthreads();
  const int wave = threadIdx.x >> 5, lane = threadIdx.x & 31;
  const int hi = lane >> 4, nn = lane & 15;
  const long long stride = (long long)gridDim.x * 64;

  for (long long row0 = (long long)blockIdx.x * 64 + wave * 16; row0 < Nn; row0 += stride) {
    v8f acc[8] = {};
    float f0 = 0.f, f1 = 0.f, f2 = 0.f, f3 = 0.f;
#pragma unroll
    for (int ks = 0; ks < 4; ks++) {
      float av[16];
      v16h a = load_afrag(x, row0, ks * 32, lane, av);
      const int kh = ks * 32 + (hi << 3);
#pragma unroll
      for (int i = 0; i < 16; i++) {
        int k = kh + ((i < 8) ? i : (i + 8));
        f0 += av[i] * vs[k * 2 + 0];
        f1 += av[i] * vs[k * 2 + 1];
        f2 += av[i] * vd[k * 2 + 0];
        f3 += av[i] * vd[k * 2 + 1];
      }
#pragma unroll
      for (int nt = 0; nt < 8; nt++) {
        v16h b = load_bfrag(wfrag, ks * 8 + nt, lane);
        acc[nt] = WMMA_F16(a, b, acc[nt]);
      }
    }
    f0 += __shfl_xor(f0, 16, 32);
    f1 += __shfl_xor(f1, 16, 32);
    f2 += __shfl_xor(f2, 16, 32);
    f3 += __shfl_xor(f3, 16, 32);
    if (lane < 16) {
      long long r = row0 + lane;
      asrc[r * 2 + 0] = f0; asrc[r * 2 + 1] = f1;
      adst[r * 2 + 0] = f2; adst[r * 2 + 1] = f3;
    }
#pragma unroll
    for (int nt = 0; nt < 8; nt++)
#pragma unroll
      for (int r = 0; r < 8; r++) {
        long long row = row0 + r + hi * 8;
        xh[row * 128 + nt * 16 + nn] = acc[nt][r];
      }
  }
}

// ---------------- K2: edge path, fused ----------------
// a_edge = edge_attr @ v_edge   (fold)
// e2 = LN(relu(edge_attr @ emlp_w + emlp_b) + edge_attr)   (WMMA f16 GEMM)
__global__ __launch_bounds__(64) void k_edge(
    const float* __restrict__ ea, const float* __restrict__ Wem,
    const float* __restrict__ vedge, const float* __restrict__ emlpb,
    const float* __restrict__ lneg, const float* __restrict__ lneb,
    float* __restrict__ aedge, float* __restrict__ e2, long long Ee) {
  __shared__ _Float16 wfrag[16384];
  __shared__ float tile[2][2048];
  __shared__ float ve[256];
  __shared__ float gsh[128], bsh[128], mbsh[128];
  __shared__ float stats[2][32];
  fill_wfrag(Wem, wfrag);
  for (int t = threadIdx.x; t < 256; t += blockDim.x) ve[t] = vedge[t];
  for (int t = threadIdx.x; t < 128; t += blockDim.x) {
    gsh[t] = lneg[t]; bsh[t] = lneb[t]; mbsh[t] = emlpb[t];
  }
  __syncthreads();
  const int wave = threadIdx.x >> 5, lane = threadIdx.x & 31;
  const int hi = lane >> 4, nn = lane & 15;
  float* tb = tile[wave];
  const long long stride = (long long)gridDim.x * 32;

  for (long long row0 = (long long)blockIdx.x * 32 + wave * 16; row0 < Ee; row0 += stride) {
    v8f acc[8] = {};
    float f0 = 0.f, f1 = 0.f;
#pragma unroll
    for (int ks = 0; ks < 4; ks++) {
      float av[16];
      v16h a = load_afrag(ea, row0, ks * 32, lane, av);
      const int kh = ks * 32 + (hi << 3);
#pragma unroll
      for (int i = 0; i < 16; i++) {
        int k = kh + ((i < 8) ? i : (i + 8));
        f0 += av[i] * ve[k * 2 + 0];
        f1 += av[i] * ve[k * 2 + 1];
      }
#pragma unroll
      for (int nt = 0; nt < 8; nt++) {
        v16h b = load_bfrag(wfrag, ks * 8 + nt, lane);
        acc[nt] = WMMA_F16(a, b, acc[nt]);
      }
    }
    f0 += __shfl_xor(f0, 16, 32);
    f1 += __shfl_xor(f1, 16, 32);
    if (lane < 16) {
      long long r = row0 + lane;
      aedge[r * 2 + 0] = f0;
      aedge[r * 2 + 1] = f1;
    }
    // relu + bias + residual -> LDS tile
#pragma unroll
    for (int nt = 0; nt < 8; nt++)
#pragma unroll
      for (int r = 0; r < 8; r++) {
        int rl = r + hi * 8;
        int col = nt * 16 + nn;
        float v = acc[nt][r] + mbsh[col];
        v = (v > 0.f) ? v : 0.f;
        v += ea[(row0 + rl) * 128 + col];
        tb[rl * 128 + col] = v;
      }
    // row stats: two lanes per row (lane, lane+16) each handle 64 cols
    {
      int r = nn, c0 = hi * 64;
      float s = 0.f, ss = 0.f;
#pragma unroll
      for (int j = 0; j < 64; j += 4) {
        float4 v4 = *(const float4*)&tb[r * 128 + c0 + j];
        s += v4.x + v4.y + v4.z + v4.w;
        ss += v4.x * v4.x + v4.y * v4.y + v4.z * v4.z + v4.w * v4.w;
      }
      s += __shfl_xor(s, 16, 32);
      ss += __shfl_xor(ss, 16, 32);
      float mean = s * (1.f / 128.f);
      float rstd = rsqrtf(ss * (1.f / 128.f) - mean * mean + 1e-5f);
      if (lane < 16) { stats[wave][r * 2] = mean; stats[wave][r * 2 + 1] = rstd; }
    }
    // coalesced LN writeback
#pragma unroll 4
    for (int r = 0; r < 16; r++) {
      float mean = stats[wave][r * 2], rstd = stats[wave][r * 2 + 1];
      int c = lane * 4;
      float4 v4 = *(const float4*)&tb[r * 128 + c];
      float4 o;
      o.x = (v4.x - mean) * rstd * gsh[c + 0] + bsh[c + 0];
      o.y = (v4.y - mean) * rstd * gsh[c + 1] + bsh[c + 1];
      o.z = (v4.z - mean) * rstd * gsh[c + 2] + bsh[c + 2];
      o.w = (v4.w - mean) * rstd * gsh[c + 3] + bsh[c + 3];
      *(float4*)&e2[(row0 + r) * 128 + c] = o;
    }
  }
}

// ---------------- K4: alpha + segment max (encoded uint atomicMax) ----------------
__global__ __launch_bounds__(256) void k_alpha(
    const float* __restrict__ asrc, const float* __restrict__ adst,
    const float* __restrict__ aedge, const int* __restrict__ src,
    const int* __restrict__ dst, float* __restrict__ alpha,
    unsigned* __restrict__ amaxK, long long E2) {
  long long t = (long long)blockIdx.x * blockDim.x + threadIdx.x;
  if (t >= E2) return;
  long long e = t >> 1;
  int h = (int)(t & 1);
  int s = src[e], d = dst[e];
  float a = asrc[s * 2 + h] + adst[d * 2 + h] + aedge[t];
  a = (a > 0.f) ? a : 0.2f * a;  // leaky_relu(0.2)
  alpha[t] = a;
  atomicMax(&amaxK[d * 2 + h], encf(a));
}

// ---------------- K5: exp + segment sum ----------------
__global__ __launch_bounds__(256) void k_expdenom(
    const float* __restrict__ alpha, const unsigned* __restrict__ amaxK,
    float* __restrict__ denom, const int* __restrict__ dst,
    float* __restrict__ exv, long long E2) {
  long long t = (long long)blockIdx.x * blockDim.x + threadIdx.x;
  if (t >= E2) return;
  long long e = t >> 1;
  int h = (int)(t & 1);
  int d = dst[e];
  unsigned u = amaxK[d * 2 + h];
  float m = (u == 0u) ? 0.f : decf(u);  // empty-segment sentinel -> 0 (ref isfinite fixup)
  float ex = __expf(alpha[t] - m);
  exv[t] = ex;
  atomicAdd(&denom[d * 2 + h], ex);
}

// ---------------- K6: weighted message scatter (one wave per edge) ----------------
__global__ __launch_bounds__(256) void k_scatter(
    const float* __restrict__ xh, const float* __restrict__ exv,
    const float* __restrict__ denom, const int* __restrict__ src,
    const int* __restrict__ dst, float* __restrict__ outacc, long long Ee) {
  long long e = (long long)blockIdx.x * 8 + (threadIdx.x >> 5);
  if (e >= Ee) return;
  int lane = threadIdx.x & 31;
  int s = src[e], d = dst[e];
  int h = lane >> 4;  // cols 0..63 -> head 0, 64..127 -> head 1
  float w = exv[e * 2 + h] / (denom[d * 2 + h] + 1e-16f);
  int c = lane * 4;
  float4 v = *(const float4*)(xh + (long long)s * 128 + c);
  float* o = outacc + (long long)d * 128 + c;
  atomicAdd(o + 0, v.x * w);
  atomicAdd(o + 1, v.y * w);
  atomicAdd(o + 2, v.z * w);
  atomicAdd(o + 3, v.w * w);
}

// ---------------- K7: node finalize, fused ----------------
// x1 = LN1(outacc + bias + x); x2 = LN2(relu(x1 @ mlp_w + mlp_b) + x1)
__global__ __launch_bounds__(64) void k_node_final(
    const float* __restrict__ xin, const float* __restrict__ outacc,
    const float* __restrict__ Wm, const float* __restrict__ nodeb,
    const float* __restrict__ l1g, const float* __restrict__ l1b,
    const float* __restrict__ mlpb, const float* __restrict__ l2g,
    const float* __restrict__ l2b, float* __restrict__ x2, long long Nn) {
  __shared__ _Float16 wfrag[16384];
  __shared__ float tile[2][2048];
  __shared__ float nb[128], g1[128], b1[128], mbsh[128], g2[128], b2[128];
  __shared__ float stats[2][32];
  fill_wfrag(Wm, wfrag);
  for (int t = threadIdx.x; t < 128; t += blockDim.x) {
    nb[t] = nodeb[t]; g1[t] = l1g[t]; b1[t] = l1b[t];
    mbsh[t] = mlpb[t]; g2[t] = l2g[t]; b2[t] = l2b[t];
  }
  __syncthreads();
  const int wave = threadIdx.x >> 5, lane = threadIdx.x & 31;
  const int hi = lane >> 4, nn = lane & 15;
  float* tb = tile[wave];
  const long long stride = (long long)gridDim.x * 32;

  for (long long row0 = (long long)blockIdx.x * 32 + wave * 16; row0 < Nn; row0 += stride) {
    // stage t = outacc + x + bias into LDS tile, compute LN1, apply in place -> x1
    {
      int r = nn, c0 = hi * 64;
      long long g = (row0 + r) * 128 + c0;
      float s = 0.f, ss = 0.f;
#pragma unroll
      for (int j = 0; j < 64; j += 4) {
        float4 o4 = *(const float4*)(outacc + g + j);
        float4 x4 = *(const float4*)(xin + g + j);
        float4 tt;
        tt.x = o4.x + x4.x + nb[c0 + j + 0];
        tt.y = o4.y + x4.y + nb[c0 + j + 1];
        tt.z = o4.z + x4.z + nb[c0 + j + 2];
        tt.w = o4.w + x4.w + nb[c0 + j + 3];
        s += tt.x + tt.y + tt.z + tt.w;
        ss += tt.x * tt.x + tt.y * tt.y + tt.z * tt.z + tt.w * tt.w;
        *(float4*)&tb[r * 128 + c0 + j] = tt;
      }
      s += __shfl_xor(s, 16, 32);
      ss += __shfl_xor(ss, 16, 32);
      float mean = s * (1.f / 128.f);
      float rstd = rsqrtf(ss * (1.f / 128.f) - mean * mean + 1e-5f);
#pragma unroll
      for (int j = 0; j < 64; j += 4) {
        float4 tt = *(const float4*)&tb[r * 128 + c0 + j];
        tt.x = (tt.x - mean) * rstd * g1[c0 + j + 0] + b1[c0 + j + 0];
        tt.y = (tt.y - mean) * rstd * g1[c0 + j + 1] + b1[c0 + j + 1];
        tt.z = (tt.z - mean) * rstd * g1[c0 + j + 2] + b1[c0 + j + 2];
        tt.w = (tt.w - mean) * rstd * g1[c0 + j + 3] + b1[c0 + j + 3];
        *(float4*)&tb[r * 128 + c0 + j] = tt;
      }
    }
    // GEMM x1 @ mlp_w from the LDS tile
    v8f acc[8] = {};
#pragma unroll
    for (int ks = 0; ks < 4; ks++) {
      v16h a = load_afrag_lds(tb, ks * 32, lane);
#pragma unroll
      for (int nt = 0; nt < 8; nt++) {
        v16h b = load_bfrag(wfrag, ks * 8 + nt, lane);
        acc[nt] = WMMA_F16(a, b, acc[nt]);
      }
    }
    // relu + bias + x1 residual, overwrite tile
#pragma unroll
    for (int nt = 0; nt < 8; nt++)
#pragma unroll
      for (int r = 0; r < 8; r++) {
        int rl = r + hi * 8;
        int col = nt * 16 + nn;
        float v = acc[nt][r] + mbsh[col];
        v = (v > 0.f) ? v : 0.f;
        v += tb[rl * 128 + col];
        tb[rl * 128 + col] = v;
      }
    // LN2 stats
    {
      int r = nn, c0 = hi * 64;
      float s = 0.f, ss = 0.f;
#pragma unroll
      for (int j = 0; j < 64; j += 4) {
        float4 v4 = *(const float4*)&tb[r * 128 + c0 + j];
        s += v4.x + v4.y + v4.z + v4.w;
        ss += v4.x * v4.x + v4.y * v4.y + v4.z * v4.z + v4.w * v4.w;
      }
      s += __shfl_xor(s, 16, 32);
      ss += __shfl_xor(ss, 16, 32);
      float mean = s * (1.f / 128.f);
      float rstd = rsqrtf(ss * (1.f / 128.f) - mean * mean + 1e-5f);
      if (lane < 16) { stats[wave][r * 2] = mean; stats[wave][r * 2 + 1] = rstd; }
    }
    // coalesced LN2 writeback
#pragma unroll 4
    for (int r = 0; r < 16; r++) {
      float mean = stats[wave][r * 2], rstd = stats[wave][r * 2 + 1];
      int c = lane * 4;
      float4 v4 = *(const float4*)&tb[r * 128 + c];
      float4 o;
      o.x = (v4.x - mean) * rstd * g2[c + 0] + b2[c + 0];
      o.y = (v4.y - mean) * rstd * g2[c + 1] + b2[c + 1];
      o.z = (v4.z - mean) * rstd * g2[c + 2] + b2[c + 2];
      o.w = (v4.w - mean) * rstd * g2[c + 3] + b2[c + 3];
      *(float4*)&x2[(row0 + r) * 128 + c] = o;
    }
  }
}

// ---------------- host launcher ----------------
extern "C" void kernel_launch(void* const* d_in, const int* in_sizes, int n_in,
                              void* d_out, int out_size, void* d_ws, size_t ws_size,
                              hipStream_t stream) {
  (void)n_in; (void)out_size; (void)ws_size;
  const float* x    = (const float*)d_in[0];
  const float* ea   = (const float*)d_in[1];
  const float* Wn   = (const float*)d_in[2];
  const float* atts = (const float*)d_in[3];
  const float* attd = (const float*)d_in[4];
  const float* We   = (const float*)d_in[5];
  const float* atte = (const float*)d_in[6];
  const float* bias = (const float*)d_in[7];
  const float* l1g  = (const float*)d_in[8];
  const float* l1b  = (const float*)d_in[9];
  const float* Wm   = (const float*)d_in[10];
  const float* mb   = (const float*)d_in[11];
  const float* Wem  = (const float*)d_in[12];
  const float* emb  = (const float*)d_in[13];
  const float* l2g  = (const float*)d_in[14];
  const float* l2b  = (const float*)d_in[15];
  const float* leg  = (const float*)d_in[16];
  const float* leb  = (const float*)d_in[17];
  const int*   ei   = (const int*)d_in[18];

  const long long N = in_sizes[0] / 128;
  const long long E = in_sizes[1] / 128;
  const int* src = ei;
  const int* dst = ei + E;

  char* ws = (char*)d_ws;
  size_t off = 0;
  auto alloc = [&](size_t bytes) -> char* {
    char* p = ws + off;
    off += (bytes + 255) & ~(size_t)255;
    return p;
  };
  float*    vsrc  = (float*)alloc(256 * 4);
  float*    vdst  = (float*)alloc(256 * 4);
  float*    vedge = (float*)alloc(256 * 4);
  float*    xh    = (float*)alloc((size_t)N * 128 * 4);
  float*    asrc  = (float*)alloc((size_t)N * 2 * 4);
  float*    adst  = (float*)alloc((size_t)N * 2 * 4);
  float*    aedge = (float*)alloc((size_t)E * 2 * 4);
  float*    alpha = (float*)alloc((size_t)E * 2 * 4);
  float*    exv   = (float*)alloc((size_t)E * 2 * 4);
  unsigned* amaxK = (unsigned*)alloc((size_t)N * 2 * 4);
  float*    denom = (float*)alloc((size_t)N * 2 * 4);
  float*    outa  = (float*)alloc((size_t)N * 128 * 4);

  float* x2 = (float*)d_out;
  float* e2 = (float*)d_out + (size_t)N * 128;

  const long long n128 = N * 128;
  const long long E2 = E * 2;

  // Capped grids: GEMM kernels grid-stride so each block amortizes its
  // 32 KB LDS weight swizzle over many row tiles.
  int gNode = (int)((N + 63) / 64);  if (gNode > 1024) gNode = 1024;
  int gEdge = (int)((E + 31) / 32);  if (gEdge > 2048) gEdge = 2048;
  int gFin  = (int)((N + 31) / 32);  if (gFin  > 1024) gFin  = 1024;

  k_fold<<<1, 128, 0, stream>>>(Wn, atts, attd, We, atte, vsrc, vdst, vedge);
  k_init<<<(int)((n128 + 255) / 256), 256, 0, stream>>>(outa, amaxK, denom, n128, N * 2);
  k_node_gemm<<<gNode, 128, 0, stream>>>(x, Wn, vsrc, vdst, xh, asrc, adst, N);
  k_edge<<<gEdge, 64, 0, stream>>>(ea, Wem, vedge, emb, leg, leb, aedge, e2, E);
  k_alpha<<<(int)((E2 + 255) / 256), 256, 0, stream>>>(asrc, adst, aedge, src, dst, alpha, amaxK, E2);
  k_expdenom<<<(int)((E2 + 255) / 256), 256, 0, stream>>>(alpha, amaxK, denom, dst, exv, E2);
  k_scatter<<<(int)((E + 7) / 8), 256, 0, stream>>>(xh, exv, denom, src, dst, outa, E);
  k_node_final<<<gFin, 64, 0, stream>>>(x, outa, Wm, bias, l1g, l1b, mb, l2g, l2b, x2, N);
}
